// GraphTransformer_82746839925405
// MI455X (gfx1250) — matrix operand order repaired and checked by reading, not compiled, not run on hardware
//
#include <hip/hip_runtime.h>
#include <stdint.h>

// ---------------------------------------------------------------------------
// GraphTransformer forward for MI455X (gfx1250, wave32, WGP).
// Dense projections: v_wmma_f32_16x16x32_bf16, one wave per 16x128 row-strip,
// B-chunk staged in LDS (bank-conflict-free padded stride), A loaded once per
// k-step; all 8 B fragments of a k-step preloaded so the 8 WMMAs issue
// back-to-back behind a single s_wait_dscnt.
// Edge attention: streaming float4 passes with float atomics.
// ---------------------------------------------------------------------------

typedef __attribute__((ext_vector_type(8)))  float  v8f;
typedef __attribute__((ext_vector_type(16))) __bf16 v16bf;

#define DIMC   128   // hidden dim D
#define HEADS  8
#define HC     16    // per-head dim C
#define LAYERS 4
#define TOUT   10
#define BPAD   136   // LDS row stride in bf16 (128 + 8): 16B-aligned, conflict-free

union Frag { v16bf v; uint4 u[2]; };

__device__ __forceinline__ __bf16 f2bf(float f) {
  unsigned u = __float_as_uint(f);
  unsigned r = (u + 0x7FFFu + ((u >> 16) & 1u)) >> 16;  // round-to-nearest-even
  unsigned short s = (unsigned short)r;
  return __builtin_bit_cast(__bf16, s);
}
// order-preserving float <-> uint for atomicMax-based segment max
__device__ __forceinline__ unsigned fenc(float f) {
  unsigned u = __float_as_uint(f);
  return (u & 0x80000000u) ? ~u : (u | 0x80000000u);
}
__device__ __forceinline__ float fdec(unsigned u) {
  return (u & 0x80000000u) ? __uint_as_float(u & 0x7FFFFFFFu)
                           : __uint_as_float(~u);
}

// ---------------------------------------------------------------------------
// WMMA GEMM:  C[M,NCOLS] = A[M,128](bf16) * B[128,NCOLS] (+bias)
// B supplied K-transposed (BT[n*128+k]).  Block = 8 waves = 128 rows x 128
// cols; the 128x128 B chunk is staged once into LDS.  Each wave: 8 f32
// accumulators (16x128 strip); per 32-deep k-step: 1 A fragment (global) +
// 8 B fragments (LDS, preloaded) -> 8 chained WMMAs.
// ---------------------------------------------------------------------------
template <int NCOLS, bool NTSTORE>
__global__ __launch_bounds__(256) void k_gemm_bf16_wmma(
    const __bf16* __restrict__ A, const __bf16* __restrict__ BT,
    const float* __restrict__ bias, float* __restrict__ C, int M) {
  __shared__ __bf16 Bsh[128 * BPAD];

  const int lane = threadIdx.x & 31;
  const int wave = threadIdx.x >> 5;
  constexpr int nChunks = NCOLS / 128;
  const int bm = blockIdx.x / nChunks;   // 128-row block
  const int nc = blockIdx.x % nChunks;   // 128-col chunk

  // ---- cooperative stage of 128x128 bf16 weight chunk into LDS ----
  {
    const uint4* src = (const uint4*)(BT + (size_t)nc * 128 * DIMC);
#pragma unroll
    for (int it = 0; it < 4; ++it) {
      int idx = threadIdx.x + it * 256;          // 1024 x 16B total
      int row = idx >> 3, c16 = idx & 7;
      *(uint4*)(Bsh + row * BPAD + c16 * 8) = src[idx];
    }
  }
  __syncthreads();

  const int tileM = bm * 8 + wave;               // wave-uniform
  if (tileM * 16 >= M) return;
  const int m0 = tileM * 16;
  const int grp = lane >> 4;                     // K half selector
  const int lo16 = lane & 15;

  const __bf16* aPtr = A + (size_t)(m0 + lo16) * DIMC;   // A row = lane%16
  const __bf16* bBase = Bsh + lo16 * BPAD + 8 * grp;
  v8f acc[8] = {};

#pragma unroll
  for (int kb = 0; kb < DIMC; kb += 32) {
    // A 16x32 frag: VGPR0..3 = K {8g..8g+7}, VGPR4..7 = K {16+8g..+7}
    Frag af;
    af.u[0] = *(const uint4*)(aPtr + kb + 8 * grp);
    af.u[1] = *(const uint4*)(aPtr + kb + 16 + 8 * grp);
    if (kb + 32 < DIMC) __builtin_prefetch(aPtr + kb + 32, 0, 0);
    // Preload all 8 B fragments for this k-step (single ds clause + wait)
    Frag bfr[8];
#pragma unroll
    for (int t = 0; t < 8; ++t) {
      const __bf16* bp = bBase + t * (16 * BPAD) + kb;
      bfr[t].u[0] = *(const uint4*)(bp);
      bfr[t].u[1] = *(const uint4*)(bp + 16);
    }
#pragma unroll
    for (int t = 0; t < 8; ++t)
      acc[t] = __builtin_amdgcn_wmma_f32_16x16x32_bf16(
          false, af.v, false, bfr[t].v, (short)0, acc[t], false, false);
  }

  // C/D layout: lane -> column, VGPR j -> rows m0+8g+j
  const int nbase = nc * 128;
#pragma unroll
  for (int t = 0; t < 8; ++t) {
    const int n = nbase + t * 16 + lo16;
    const float bv = bias ? bias[n] : 0.0f;
    float* cp = C + (size_t)(m0 + 8 * grp) * NCOLS + n;
#pragma unroll
    for (int j = 0; j < 8; ++j) {
      float val = acc[t][j] + bv;
      if (NTSTORE) __builtin_nontemporal_store(val, cp + j * NCOLS);
      else         cp[j * NCOLS] = val;
    }
  }
}

// ---------------------------------------------------------------------------
// Encoders
// ---------------------------------------------------------------------------
__global__ void k_encode_nodes(const float* __restrict__ x,
                               const float* __restrict__ w,
                               const float* __restrict__ b,
                               float* __restrict__ h, __bf16* __restrict__ hbf,
                               int n) {
  int i = blockIdx.x * blockDim.x + threadIdx.x;
  if (i >= n * DIMC) return;
  int node = i >> 7, d = i & 127;
  float s = b[d];
  const float* xr = x + node * 9;
#pragma unroll
  for (int j = 0; j < 9; ++j) s += xr[j] * w[j * DIMC + d];
  s = fmaxf(s, 0.0f);
  h[i] = s;
  hbf[i] = f2bf(s);
}

__global__ void k_encode_edges(const float* __restrict__ ea,
                               const float* __restrict__ w,
                               const float* __restrict__ b,
                               __bf16* __restrict__ eabf, int e) {
  int i = blockIdx.x * blockDim.x + threadIdx.x;
  if (i >= e * DIMC) return;
  int ed = i >> 7, d = i & 127;
  float s = b[d];
  const float* ar = ea + ed * 3;
#pragma unroll
  for (int j = 0; j < 3; ++j) s += ar[j] * w[j * DIMC + d];
  eabf[i] = f2bf(s);
}

// ---------------------------------------------------------------------------
// One-time weight prep: bf16 convert + K-transpose.
// WcatT: [L][512][128] columns = [Wq|Wk|Wv|Wskip]; WeT: [L][128][128]; bcat [L][512].
// ---------------------------------------------------------------------------
__global__ void k_prep_weights(const float* Wq, const float* bq,
                               const float* Wk, const float* bk,
                               const float* Wv, const float* bv,
                               const float* Wsk, const float* bsk,
                               const float* We,
                               __bf16* WcatT, __bf16* WeT, float* bcat) {
  const int per = DIMC * DIMC;
  int idx = blockIdx.x * blockDim.x + threadIdx.x;
  if (idx >= LAYERS * 5 * per) return;
  int l = idx / (5 * per);
  int sel = (idx / per) % 5;
  int r = idx % per;
  int k = r / DIMC, n = r % DIMC;
  const float* W; const float* B = nullptr;
  switch (sel) {
    case 0: W = Wq;  B = bq;  break;
    case 1: W = Wk;  B = bk;  break;
    case 2: W = Wv;  B = bv;  break;
    case 3: W = Wsk; B = bsk; break;
    default: W = We; break;
  }
  float wv = W[(size_t)l * per + k * DIMC + n];
  if (sel < 4) {
    WcatT[((size_t)l * 512 + sel * DIMC + n) * DIMC + k] = f2bf(wv);
    if (k == 0) bcat[l * 512 + sel * DIMC + n] = B[l * DIMC + n];
  } else {
    WeT[(size_t)l * per + (size_t)n * DIMC + k] = f2bf(wv);
  }
}

// ---------------------------------------------------------------------------
// Per-layer scratch init
// ---------------------------------------------------------------------------
__global__ void k_layer_init(float* agg, unsigned* menc, float* denom, int n) {
  int i = blockIdx.x * blockDim.x + threadIdx.x;
  if (i < n * DIMC) agg[i] = 0.0f;
  if (i < n * HEADS) { menc[i] = 0u; denom[i] = 0.0f; }
}

// ---------------------------------------------------------------------------
// Edge attention: raw score + segment max / exp + segment sum / scatter
// qkvs row layout: [q(0:128) | k(128:256) | v(256:384) | x_r(384:512)]
// All head slices are 64B-aligned -> float4 gathers.
// ---------------------------------------------------------------------------
__global__ void k_alpha_max(const int* __restrict__ src, const int* __restrict__ dst,
                            const float* __restrict__ qkvs, const float* __restrict__ eF,
                            float* __restrict__ alpha, unsigned* __restrict__ menc,
                            int E_) {
  int idx = blockIdx.x * blockDim.x + threadIdx.x;
  if (idx >= E_ * HEADS) return;
  int e = idx >> 3, hd = idx & 7;
  int s = src[e], d = dst[e];
  const float4* q  = (const float4*)(qkvs + (size_t)d * 512 + hd * HC);
  const float4* k  = (const float4*)(qkvs + (size_t)s * 512 + DIMC + hd * HC);
  const float4* ee = (const float4*)(eF + (size_t)e * DIMC + hd * HC);
  float a = 0.0f;
#pragma unroll
  for (int c = 0; c < 4; ++c) {
    float4 qv = q[c], kv = k[c], ev = ee[c];
    a += qv.x * (kv.x + ev.x) + qv.y * (kv.y + ev.y) +
         qv.z * (kv.z + ev.z) + qv.w * (kv.w + ev.w);
  }
  a *= 0.25f;  // 1/sqrt(16)
  alpha[idx] = a;
  atomicMax(&menc[d * HEADS + hd], fenc(a));
}

__global__ void k_alpha_exp(const int* __restrict__ dst,
                            float* __restrict__ alpha,
                            const unsigned* __restrict__ menc,
                            float* __restrict__ denom, int E_) {
  int idx = blockIdx.x * blockDim.x + threadIdx.x;
  if (idx >= E_ * HEADS) return;
  int e = idx >> 3, hd = idx & 7;
  int d = dst[e];
  float v = __expf(alpha[idx] - fdec(menc[d * HEADS + hd]));
  alpha[idx] = v;
  atomicAdd(&denom[d * HEADS + hd], v);
}

__global__ void k_scatter(const int* __restrict__ src, const int* __restrict__ dst,
                          const float* __restrict__ qkvs, const float* __restrict__ eF,
                          const float* __restrict__ alpha, const float* __restrict__ denom,
                          float* __restrict__ agg, int E_) {
  int idx = blockIdx.x * blockDim.x + threadIdx.x;
  if (idx >= E_ * HEADS) return;
  int e = idx >> 3, hd = idx & 7;
  int s = src[e], d = dst[e];
  float w = alpha[idx] / (denom[d * HEADS + hd] + 1e-16f);
  const float4* vv = (const float4*)(qkvs + (size_t)s * 512 + 2 * DIMC + hd * HC);
  const float4* ee = (const float4*)(eF + (size_t)e * DIMC + hd * HC);
  float* out = agg + (size_t)d * DIMC + hd * HC;
#pragma unroll
  for (int c = 0; c < 4; ++c) {
    float4 v4 = vv[c], e4 = ee[c];
    atomicAdd(&out[c * 4 + 0], (v4.x + e4.x) * w);
    atomicAdd(&out[c * 4 + 1], (v4.y + e4.y) * w);
    atomicAdd(&out[c * 4 + 2], (v4.z + e4.z) * w);
    atomicAdd(&out[c * 4 + 3], (v4.w + e4.w) * w);
  }
}

// ---------------------------------------------------------------------------
// Beta gate + residual + LayerNorm; one wave32 per node, 4 cols per lane.
// ---------------------------------------------------------------------------
__global__ void k_beta_ln(const float* __restrict__ agg, const float* __restrict__ qkvs,
                          const float* __restrict__ Wb,
                          const float* __restrict__ lng, const float* __restrict__ lnb,
                          float* __restrict__ h, __bf16* __restrict__ hbf, int n) {
  int lane = threadIdx.x & 31;
  int node = blockIdx.x * (blockDim.x >> 5) + (threadIdx.x >> 5);
  if (node >= n) return;
  int col0 = lane * 4;
  float4 o4 = *(const float4*)(agg + (size_t)node * DIMC + col0);
  float4 x4 = *(const float4*)(qkvs + (size_t)node * 512 + 3 * DIMC + col0);
  float o[4]  = {o4.x, o4.y, o4.z, o4.w};
  float xr[4] = {x4.x, x4.y, x4.z, x4.w};

  float p = 0.0f;
#pragma unroll
  for (int t = 0; t < 4; ++t) {
    int c = col0 + t;
    p += o[t] * Wb[c] + xr[t] * Wb[DIMC + c] + (o[t] - xr[t]) * Wb[2 * DIMC + c];
  }
#pragma unroll
  for (int off = 16; off; off >>= 1) p += __shfl_xor(p, off);
  float beta = 1.0f / (1.0f + __expf(-p));

  const float* hrow = h + (size_t)node * DIMC;
  float hn[4]; float ssum = 0.0f;
#pragma unroll
  for (int t = 0; t < 4; ++t) {
    hn[t] = beta * xr[t] + (1.0f - beta) * o[t] + hrow[col0 + t];  // + residual
    ssum += hn[t];
  }
#pragma unroll
  for (int off = 16; off; off >>= 1) ssum += __shfl_xor(ssum, off);
  float mu = ssum * (1.0f / DIMC);
  float vs = 0.0f;
#pragma unroll
  for (int t = 0; t < 4; ++t) { float dd = hn[t] - mu; vs += dd * dd; }
#pragma unroll
  for (int off = 16; off; off >>= 1) vs += __shfl_xor(vs, off);
  float inv = rsqrtf(vs * (1.0f / DIMC) + 1e-5f);
#pragma unroll
  for (int t = 0; t < 4; ++t) {
    int c = col0 + t;
    float y = (hn[t] - mu) * inv * lng[c] + lnb[c];
    h[(size_t)node * DIMC + c] = y;
    hbf[(size_t)node * DIMC + c] = f2bf(y);
  }
}

// ---------------------------------------------------------------------------
// Mean pool over graphs + MLP head
// ---------------------------------------------------------------------------
__global__ void k_pool_init(float* pooled, float* cnt, int g) {
  int i = blockIdx.x * blockDim.x + threadIdx.x;
  if (i < g * DIMC) pooled[i] = 0.0f;
  if (i < g) cnt[i] = 0.0f;
}

__global__ void k_pool(const float* __restrict__ h, const int* __restrict__ batch,
                       float* __restrict__ pooled, float* __restrict__ cnt, int n) {
  int i = blockIdx.x * blockDim.x + threadIdx.x;
  if (i >= n * DIMC) return;
  int node = i >> 7, d = i & 127;
  int g = batch[node];
  atomicAdd(&pooled[(size_t)g * DIMC + d], h[i]);
  if (d == 0) atomicAdd(&cnt[g], 1.0f);
}

__global__ void k_head(const float* __restrict__ pooled, const float* __restrict__ cnt,
                       const float* __restrict__ w1, const float* __restrict__ b1,
                       const float* __restrict__ w2, const float* __restrict__ b2,
                       float* __restrict__ out) {
  __shared__ float sp[DIMC];
  __shared__ float sh[64];
  int g = blockIdx.x, t = threadIdx.x;  // 64 threads
  float inv = 1.0f / fmaxf(cnt[g], 1.0f);
  sp[t]      = pooled[(size_t)g * DIMC + t] * inv;
  sp[t + 64] = pooled[(size_t)g * DIMC + t + 64] * inv;
  __syncthreads();
  float a = b1[t];
#pragma unroll 8
  for (int d = 0; d < DIMC; ++d) a += sp[d] * w1[d * 64 + t];
  sh[t] = fmaxf(a, 0.0f);
  __syncthreads();
  if (t < TOUT) {
    float s = b2[t];
#pragma unroll 8
    for (int c = 0; c < 64; ++c) s += sh[c] * w2[c * TOUT + t];
    out[g * TOUT + t] = s;
  }
}

// ---------------------------------------------------------------------------
extern "C" void kernel_launch(void* const* d_in, const int* in_sizes, int n_in,
                              void* d_out, int out_size, void* d_ws, size_t ws_size,
                              hipStream_t stream) {
  (void)n_in; (void)ws_size;
  const float* x         = (const float*)d_in[0];
  const int*   edge_idx  = (const int*)d_in[1];
  const float* edge_attr = (const float*)d_in[2];
  const int*   batch     = (const int*)d_in[3];
  const float* node_w = (const float*)d_in[4];
  const float* node_b = (const float*)d_in[5];
  const float* edge_w = (const float*)d_in[6];
  const float* edge_b = (const float*)d_in[7];
  const float* Wq = (const float*)d_in[8];   const float* bq = (const float*)d_in[9];
  const float* Wk = (const float*)d_in[10];  const float* bk = (const float*)d_in[11];
  const float* Wv = (const float*)d_in[12];  const float* bv = (const float*)d_in[13];
  const float* We = (const float*)d_in[14];
  const float* Wsk = (const float*)d_in[15]; const float* bsk = (const float*)d_in[16];
  const float* Wbeta = (const float*)d_in[17];
  const float* ln_g = (const float*)d_in[18]; const float* ln_b = (const float*)d_in[19];
  const float* h1_w = (const float*)d_in[20]; const float* h1_b = (const float*)d_in[21];
  const float* h2_w = (const float*)d_in[22]; const float* h2_b = (const float*)d_in[23];
  float* out = (float*)d_out;

  const int N = in_sizes[0] / 9;     // 20000 (multiple of 16)
  const int E = in_sizes[1] / 2;     // 320000 (multiple of 16)
  const int G = out_size / TOUT;     // 128
  const int* srcI = edge_idx;
  const int* dstI = edge_idx + E;

  // workspace carve-out (256B aligned chunks)
  char* wp = (char*)d_ws;
  auto alloc = [&](size_t bytes) {
    char* p = wp; wp += (bytes + 255) & ~(size_t)255; return p;
  };
  float*    h     = (float*)alloc((size_t)N * DIMC * 4);
  __bf16*   hbf   = (__bf16*)alloc((size_t)N * DIMC * 2);
  float*    qkvs  = (float*)alloc((size_t)N * 512 * 4);
  __bf16*   eabf  = (__bf16*)alloc((size_t)E * DIMC * 2);
  float*    eF    = (float*)alloc((size_t)E * DIMC * 4);
  __bf16*   WcatT = (__bf16*)alloc((size_t)LAYERS * 512 * DIMC * 2);
  __bf16*   WeT   = (__bf16*)alloc((size_t)LAYERS * DIMC * DIMC * 2);
  float*    bcat  = (float*)alloc((size_t)LAYERS * 512 * 4);
  float*    alpha = (float*)alloc((size_t)E * HEADS * 4);
  unsigned* menc  = (unsigned*)alloc((size_t)N * HEADS * 4);
  float*    denom = (float*)alloc((size_t)N * HEADS * 4);
  float*    agg   = (float*)alloc((size_t)N * DIMC * 4);
  float*    pooled= (float*)alloc((size_t)G * DIMC * 4);
  float*    cnt   = (float*)alloc((size_t)G * 4);

  const int TB = 256;
  k_encode_nodes<<<(N * DIMC + TB - 1) / TB, TB, 0, stream>>>(x, node_w, node_b, h, hbf, N);
  k_encode_edges<<<(E * DIMC + TB - 1) / TB, TB, 0, stream>>>(edge_attr, edge_w, edge_b, eabf, E);
  k_prep_weights<<<(LAYERS * 5 * DIMC * DIMC + TB - 1) / TB, TB, 0, stream>>>(
      Wq, bq, Wk, bk, Wv, bv, Wsk, bsk, We, WcatT, WeT, bcat);

  const int rowTilesN = N / 16, rowTilesE = E / 16;
  const int blocksQ = ((rowTilesN + 7) / 8) * (512 / 128);
  const int blocksE = ((rowTilesE + 7) / 8) * (128 / 128);

  for (int l = 0; l < LAYERS; ++l) {
    k_gemm_bf16_wmma<512, false><<<blocksQ, 256, 0, stream>>>(
        hbf, WcatT + (size_t)l * 512 * DIMC, bcat + l * 512, qkvs, N);
    k_gemm_bf16_wmma<128, true><<<blocksE, 256, 0, stream>>>(
        eabf, WeT + (size_t)l * DIMC * DIMC, nullptr, eF, E);
    k_layer_init<<<(N * DIMC + TB - 1) / TB, TB, 0, stream>>>(agg, menc, denom, N);
    k_alpha_max<<<(E * HEADS + TB - 1) / TB, TB, 0, stream>>>(srcI, dstI, qkvs, eF, alpha, menc, E);
    k_alpha_exp<<<(E * HEADS + TB - 1) / TB, TB, 0, stream>>>(dstI, alpha, menc, denom, E);
    k_scatter<<<(E * HEADS + TB - 1) / TB, TB, 0, stream>>>(srcI, dstI, qkvs, eF, alpha, denom, agg, E);
    k_beta_ln<<<(N + 7) / 8, 256, 0, stream>>>(
        agg, qkvs, Wbeta + l * 3 * DIMC, ln_g + l * DIMC, ln_b + l * DIMC, h, hbf, N);
  }

  k_pool_init<<<(G * DIMC + TB - 1) / TB, TB, 0, stream>>>(pooled, cnt, G);
  k_pool<<<(N * DIMC + TB - 1) / TB, TB, 0, stream>>>(h, batch, pooled, cnt, N);
  k_head<<<G, 64, 0, stream>>>(pooled, cnt, h1_w, h1_b, h2_w, h2_b, out);
}
